// CorrelationNet_8607114461662
// MI455X (gfx1250) — compile-verified, matrix operand order checked by reading, not run on hardware
//
#include <hip/hip_runtime.h>
#include <hip/hip_bf16.h>

typedef __attribute__((ext_vector_type(16))) _Float16 v16h;
typedef __attribute__((ext_vector_type(8)))  _Float16 v8h;
typedef __attribute__((ext_vector_type(8)))  float    v8f;
typedef __attribute__((ext_vector_type(4)))  unsigned int u32x4;
typedef __attribute__((ext_vector_type(8)))  unsigned int u32x8;

#define B_  8
#define C_  256
#define H_  96
#define W_  96
#define S_  81       // (2*4+1)^2 correlation channels
#define P_  128      // planes
#define HW_ (H_*W_)  // 9216
#define NN_ (B_*HW_) // 73728 elements per channel for BN stats

// ---------------------------------------------------------------------------
// Correlation: corr[b,s,y,x] = mean_c img1[b,c,y,x] * img2[b,c,y+dy,x+dx]
// ---------------------------------------------------------------------------
__global__ __launch_bounds__(256)
void corr_kernel(const float* __restrict__ img1, const float* __restrict__ img2,
                 _Float16* __restrict__ corr_h)
{
    __shared__ float s1[32*32];     // [c][x]
    __shared__ float s2[32*9*40];   // [c][r=dy+4][col = x_local + dx + 4]
    int blk = blockIdx.x;
    int xt  = blk % 3;
    int y   = (blk/3) % H_;
    int b   = blk / (3*H_);
    int x0  = xt*32;
    int tid = threadIdx.x;
    int xl  = tid & 31;
    int sg  = tid >> 5;   // 0..7

    int dys[11], dxs[11]; int ns = 0;
    #pragma unroll
    for (int j = 0; j < 11; ++j) {
        int s = sg + 8*j;
        if (s < S_) { dys[j] = s/9 - 4; dxs[j] = s%9 - 4; ns = j+1; }
        else        { dys[j] = 0;       dxs[j] = 0; }
    }
    float acc[11];
    #pragma unroll
    for (int j = 0; j < 11; ++j) acc[j] = 0.f;

    for (int c0 = 0; c0 < C_; c0 += 32) {
        __syncthreads();
        for (int e = tid; e < 32*32; e += 256) {
            int c = e >> 5, x = e & 31;
            s1[e] = img1[(((size_t)b*C_ + (c0+c))*H_ + y)*W_ + x0 + x];
        }
        for (int e = tid; e < 32*9*40; e += 256) {
            int col = e % 40; int r = (e/40) % 9; int c = e/360;
            int yy = y + r - 4;
            int xx = x0 + col - 4;
            float v = 0.f;
            if (yy >= 0 && yy < H_ && xx >= 0 && xx < W_)
                v = img2[(((size_t)b*C_ + (c0+c))*H_ + yy)*W_ + xx];
            s2[e] = v;
        }
        __syncthreads();
        for (int c = 0; c < 32; ++c) {
            float v1 = s1[c*32 + xl];
            #pragma unroll
            for (int j = 0; j < 11; ++j) {
                if (j < ns)
                    acc[j] += v1 * s2[(c*9 + (dys[j]+4))*40 + (xl + dxs[j] + 4)];
            }
        }
    }
    const float inv = 1.0f / (float)C_;
    for (int j = 0; j < ns; ++j) {
        int s = sg + 8*j;
        corr_h[(((size_t)b*S_ + s)*H_ + y)*W_ + x0 + xl] = (_Float16)(acc[j]*inv);
    }
}

// ---------------------------------------------------------------------------
// Weight swizzle: f32 OIHW -> f16 fragment-major layout: within chunk kc the
// 4096 halves are [lh][n][idx16] so a lane's B fragment is contiguous 32 B.
// ---------------------------------------------------------------------------
__global__ __launch_bounds__(256)
void wswz_kernel(const float* __restrict__ src, _Float16* __restrict__ dst,
                 int Kreal, int total)
{
    int t = blockIdx.x*blockDim.x + threadIdx.x;
    if (t >= total) return;
    int idx = t & 15;
    int n   = (t >> 4) & 127;
    int lh  = (t >> 11) & 1;
    int kc  = t >> 12;
    int K = kc*32 + ((idx >= 8) ? 16 : 0) + lh*8 + (idx & 7);
    float v = (K < Kreal) ? src[(size_t)n*Kreal + K] : 0.f;
    dst[t] = (_Float16)v;
}

// ---------------------------------------------------------------------------
// TDM: DMA one 8 KB weight chunk (4096 f16, 1-D tile) from global -> LDS.
// D# per ISA ch.8: group0 = {count=1, lds_addr, global_addr, type=2},
// group1 = {data_size=2B, tensor_dim0=tile_dim0=4096, stride=4096}.
// Groups 2/3 omitted (NULL) => <=2D tensor. Issued by one wave; completion
// tracked with TENSORcnt.
// ---------------------------------------------------------------------------
__device__ __forceinline__ void tdm_load_chunk(const _Float16* gsrc, unsigned lds_off)
{
    union { u32x4 v; unsigned w[4]; } g0;
    union { u32x8 v; unsigned w[8]; } g1;
    unsigned long long ga = (unsigned long long)(size_t)gsrc;
    g0.w[0] = 1u;                                   // count=1, user mode
    g0.w[1] = lds_off;                              // lds_addr (bytes)
    g0.w[2] = (unsigned)ga;                         // global_addr[31:0]
    g0.w[3] = (unsigned)(ga >> 32) | (2u << 30);    // global_addr[56:32] | type=2
    g1.w[0] = 1u << 16;                             // data_size=1 (2 bytes)
    g1.w[1] = (4096u & 0xFFFFu) << 16;              // tensor_dim0[15:0] @ bits63:48
    g1.w[2] = (4096u >> 16) | (1u << 16);           // tensor_dim0[31:16], tensor_dim1=1
    g1.w[3] = 4096u << 16;                          // tile_dim0 @ bits127:112
    g1.w[4] = 0;                                    // tile_dim1=0, tile_dim2=0 (unused)
    g1.w[5] = 4096u;                                // tensor_dim0_stride lo
    g1.w[6] = 0;
    g1.w[7] = 0;
    asm volatile("tensor_load_to_lds %0, %1" :: "s"(g0.v), "s"(g1.v) : "memory");
}

// ---------------------------------------------------------------------------
// Implicit-GEMM conv via v_wmma_f32_16x16x32_f16.
// Block = 256 threads = 8 waves; block tile M=64 pixels (4 y-rows x 16 x)
// by N=128 planes. Weights stream in via TDM (double-buffered in LDS,
// TENSORcnt-pipelined); A tile (64x32 f16 im2col) staged in LDS; each wave
// runs 4 independent WMMA chains reusing one B fragment.
// ---------------------------------------------------------------------------
__global__ __launch_bounds__(256)
void conv_wmma_kernel(const _Float16* __restrict__ src,   // f16 activations
                      const _Float16* __restrict__ wswz,  // swizzled weights
                      float* __restrict__ out,            // [B][128][H][W]
                      int Cin, int taps3, int Kreal, int nKc)
{
    __shared__ __align__(32) _Float16 Alds[64*32];     // 4 KB
    __shared__ __align__(32) _Float16 Wlds[2][4096];   // 2 x 8 KB ping-pong
    int blk = blockIdx.x;
    int xt  = blk % (W_/16);                 // 6
    int yt  = (blk/(W_/16)) % (H_/4);        // 24
    int b   = blk / ((W_/16)*(H_/4));
    int x0  = xt*16;
    int y0  = yt*4;
    int tid  = threadIdx.x;
    int lane = tid & 31;
    int wave = tid >> 5;       // 0..7 -> N subtile
    int mrow = lane & 15;
    int lh   = lane >> 4;

    // LDS byte offsets of the two weight buffers (low 32 bits of the flat
    // shared address == group-segment offset per the aperture mapping).
    unsigned wbase = (unsigned)(size_t)&Wlds[0][0];
    bool lead = (wave == 0);

    // Prologue: DMA chunk 0 into buffer 0.
    if (lead) tdm_load_chunk(wswz, wbase);

    v8f acc[4];
    #pragma unroll
    for (int mt = 0; mt < 4; ++mt) acc[mt] = (v8f){};

    for (int kc = 0; kc < nKc; ++kc) {
        __syncthreads();   // prior iteration done reading Wlds[(kc+1)&1]
        // im2col gather of A tile: 2048 halves, 8 per thread (coalesced in x).
        #pragma unroll
        for (int e8 = 0; e8 < 8; ++e8) {
            int e  = e8*256 + tid;
            int m  = e & 63;           // my = m>>4, mx = m&15
            int kk = e >> 6;           // 0..31
            int kg = kc*32 + kk;
            _Float16 v = (_Float16)0.f;
            if (kg < Kreal) {
                int y = y0 + (m >> 4);
                int x = x0 + (m & 15);
                if (taps3) {
                    int cch = kg/9, r = kg%9;
                    int yy = y + r/3 - 1;
                    int xx = x + (r%3) - 1;
                    if (yy >= 0 && yy < H_ && xx >= 0 && xx < W_)
                        v = src[(((size_t)b*Cin + cch)*H_ + yy)*W_ + xx];
                } else {
                    v = src[(((size_t)b*Cin + kg)*H_ + y)*W_ + x];
                }
            }
            Alds[m*32 + kk] = v;
        }
        // Pipeline the weight DMA: issue next chunk, wait for current.
        if (lead) {
            if (kc + 1 < nKc) {
                tdm_load_chunk(wswz + (size_t)(kc+1)*4096, wbase + ((kc+1)&1)*8192u);
                __builtin_amdgcn_s_wait_tensorcnt((short)1);  // chunk kc landed
            } else {
                __builtin_amdgcn_s_wait_tensorcnt((short)0);
            }
        }
        __syncthreads();   // A tile + W chunk kc visible to all waves
        // B fragment from LDS: contiguous 32 B.
        const _Float16* wch = &Wlds[kc & 1][0];
        union { v16h v; v8h h[2]; } bf;
        bf.h[0] = *(const v8h*)&wch[((lh*128) + wave*16 + mrow)*16];
        bf.h[1] = *(const v8h*)&wch[((lh*128) + wave*16 + mrow)*16 + 8];
        // 4 independent WMMA chains reusing bfrag.
        #pragma unroll
        for (int mt = 0; mt < 4; ++mt) {
            union { v16h v; v8h h[2]; } a;
            a.h[0] = *(const v8h*)&Alds[(mt*16 + mrow)*32 + lh*8];
            a.h[1] = *(const v8h*)&Alds[(mt*16 + mrow)*32 + 16 + lh*8];
            acc[mt] = __builtin_amdgcn_wmma_f32_16x16x32_f16(
                          false, a.v, false, bf.v, (short)0, acc[mt], false, false);
        }
    }
    // D: lane holds 8 consecutive x for each M-tile -> one v8f (2x b128) store.
    int n = wave*16 + mrow;
    #pragma unroll
    for (int mt = 0; mt < 4; ++mt) {
        int y = y0 + mt;
        size_t base = (((size_t)b*P_ + n)*H_ + y)*W_ + x0 + lh*8;
        *(v8f*)&out[base] = acc[mt];
    }
}

// ---------------------------------------------------------------------------
// BN statistics: one block per channel; mean + rsqrt(var+eps).
// ---------------------------------------------------------------------------
__global__ __launch_bounds__(256)
void bnstats_kernel(const float* __restrict__ x, float* __restrict__ stats)
{
    __shared__ float rs[256], rq[256];
    int c = blockIdx.x;
    int tid = threadIdx.x;
    float s = 0.f, q = 0.f;
    for (int b = 0; b < B_; ++b) {
        const float* p = x + ((size_t)b*P_ + c)*HW_;
        for (int i = tid; i < HW_; i += 256) { float v = p[i]; s += v; q += v*v; }
    }
    rs[tid] = s; rq[tid] = q;
    __syncthreads();
    for (int st = 128; st > 0; st >>= 1) {
        if (tid < st) { rs[tid] += rs[tid+st]; rq[tid] += rq[tid+st]; }
        __syncthreads();
    }
    if (tid == 0) {
        float mean = rs[0] / (float)NN_;
        float var  = rq[0] / (float)NN_ - mean*mean;
        stats[2*c]   = mean;
        stats[2*c+1] = rsqrtf(var + 1e-5f);
    }
}

// bn1 + relu -> f16 activations for conv2
__global__ __launch_bounds__(256)
void act1_kernel(const float* __restrict__ x, const float* __restrict__ stats,
                 const float* __restrict__ g, const float* __restrict__ bta,
                 _Float16* __restrict__ outh, int n)
{
    int i = blockIdx.x*blockDim.x + threadIdx.x;
    if (i >= n) return;
    int c = (i / HW_) % P_;
    float v = (x[i] - stats[2*c]) * stats[2*c+1] * g[c] + bta[c];
    v = v > 0.f ? v : 0.f;
    outh[i] = (_Float16)v;
}

// relu(bn2(out2) + bnd(outd)) -> f32 output
__global__ __launch_bounds__(256)
void final_kernel(const float* __restrict__ x2, const float* __restrict__ xd,
                  const float* __restrict__ s2, const float* __restrict__ sd,
                  const float* __restrict__ g2, const float* __restrict__ b2,
                  const float* __restrict__ gd, const float* __restrict__ bd,
                  float* __restrict__ out, int n)
{
    int i = blockIdx.x*blockDim.x + threadIdx.x;
    if (i >= n) return;
    int c = (i / HW_) % P_;
    float a = (x2[i] - s2[2*c]) * s2[2*c+1] * g2[c] + b2[c];
    float d = (xd[i] - sd[2*c]) * sd[2*c+1] * gd[c] + bd[c];
    float v = a + d;
    out[i] = v > 0.f ? v : 0.f;
}

extern "C" void kernel_launch(void* const* d_in, const int* in_sizes, int n_in,
                              void* d_out, int out_size, void* d_ws, size_t ws_size,
                              hipStream_t stream)
{
    (void)in_sizes; (void)n_in; (void)out_size; (void)ws_size;
    const float* img1   = (const float*)d_in[0];
    const float* img2   = (const float*)d_in[1];
    const float* conv1w = (const float*)d_in[2];
    const float* bn1g   = (const float*)d_in[3];
    const float* bn1b   = (const float*)d_in[4];
    const float* conv2w = (const float*)d_in[5];
    const float* bn2g   = (const float*)d_in[6];
    const float* bn2b   = (const float*)d_in[7];
    const float* downw  = (const float*)d_in[8];
    const float* bndg   = (const float*)d_in[9];
    const float* bndb   = (const float*)d_in[10];
    float* out = (float*)d_out;

    char* ws = (char*)d_ws;
    size_t off = 0;
    auto take = [&](size_t bytes) -> void* {
        void* p = ws + off;
        off += (bytes + 255) & ~((size_t)255);
        return p;
    };
    _Float16* corr_h = (_Float16*)take((size_t)B_*S_*HW_*2);   // 11.9 MB
    _Float16* w1s    = (_Float16*)take((size_t)23*4096*2);
    _Float16* wds    = (_Float16*)take((size_t)3*4096*2);
    _Float16* w2s    = (_Float16*)take((size_t)36*4096*2);
    float*    out1   = (float*)take((size_t)B_*P_*HW_*4);      // 37.7 MB
    float*    outd   = (float*)take((size_t)B_*P_*HW_*4);
    float*    out2   = (float*)take((size_t)B_*P_*HW_*4);
    _Float16* act1h  = (_Float16*)take((size_t)B_*P_*HW_*2);   // 18.9 MB
    float*    st1    = (float*)take(P_*2*sizeof(float));
    float*    st2    = (float*)take(P_*2*sizeof(float));
    float*    std_   = (float*)take(P_*2*sizeof(float));

    // weight swizzle (small)
    wswz_kernel<<<(23*4096 + 255)/256, 256, 0, stream>>>(conv1w, w1s, 729, 23*4096);
    wswz_kernel<<<(3*4096  + 255)/256, 256, 0, stream>>>(downw,  wds, 81,  3*4096);
    wswz_kernel<<<(36*4096 + 255)/256, 256, 0, stream>>>(conv2w, w2s, 1152, 36*4096);

    // correlation
    corr_kernel<<<B_*H_*3, 256, 0, stream>>>(img1, img2, corr_h);

    // conv1 (3x3, K=729) and down (1x1, K=81) from corr
    const int convGrid = B_*(H_/4)*(W_/16);   // 1152
    conv_wmma_kernel<<<convGrid, 256, 0, stream>>>(corr_h, w1s, out1, S_, 1, 729, 23);
    conv_wmma_kernel<<<convGrid, 256, 0, stream>>>(corr_h, wds, outd, S_, 0, 81, 3);

    // bn1 + relu -> f16
    bnstats_kernel<<<P_, 256, 0, stream>>>(out1, st1);
    const int ntot = B_*P_*HW_;
    act1_kernel<<<(ntot + 255)/256, 256, 0, stream>>>(out1, st1, bn1g, bn1b, act1h, ntot);

    // conv2 (3x3, K=1152)
    conv_wmma_kernel<<<convGrid, 256, 0, stream>>>(act1h, w2s, out2, P_, 1, 1152, 36);

    // bn2, bnd, fused residual relu
    bnstats_kernel<<<P_, 256, 0, stream>>>(out2, st2);
    bnstats_kernel<<<P_, 256, 0, stream>>>(outd, std_);
    final_kernel<<<(ntot + 255)/256, 256, 0, stream>>>(out2, outd, st2, std_,
                                                       bn2g, bn2b, bndg, bndb, out, ntot);
}
